// SSMSpectralLayer_68092411510854
// MI455X (gfx1250) — matrix operand order repaired and checked by reading, not compiled, not run on hardware
//
#include <hip/hip_runtime.h>
#include <math.h>

// ---------------------------------------------------------------------------
// Problem constants
// ---------------------------------------------------------------------------
#define BATCH   2
#define SEQL    1024
#define DMODEL  1024
#define STATE   16
#define CONVK   4
#define DTR     8
#define INNER   2048            // DMODEL * EXPAND
#define NI2     4096            // 2*INNER
#define BL      2048            // BATCH*SEQL
#define BE      4096            // BATCH*INNER
#define NFFT    513             // L/2+1
#define FHALF   576             // NFFT padded (cos block half-width)
#define NF2P    1152            // 2*FHALF, multiple of 128

typedef __attribute__((ext_vector_type(16))) __bf16         v16bf;
typedef __attribute__((ext_vector_type(16))) unsigned short v16u;
typedef __attribute__((ext_vector_type(8)))  float          v8f;

__device__ __forceinline__ unsigned short f2bf(float f) {
  union { float f; unsigned u; } v; v.f = f;
  unsigned r = v.u + 0x7FFFu + ((v.u >> 16) & 1u);   // round-to-nearest-even
  return (unsigned short)(r >> 16);
}
__device__ __forceinline__ float silu(float x) { return x / (1.f + __expf(-x)); }

// ---------------------------------------------------------------------------
// WMMA fragment-order (swizzled) layouts, contiguous k order per lane half, same for A and B.
// A (MxK): tile (mt,kt) of 16x32; lane = hi*16 + (m&15); per-lane 16 halves:
//   K = kt*32 + 16*hi + e
// B (KxN): tile (kt,nt) of 32x16; lane = hi*16 + (n&15); per-lane 16 halves:
//   K = kt*32 + 16*hi + e
// Chunks of 16 halves (32B) stored contiguously per lane -> b128-pair loads.
// ---------------------------------------------------------------------------
__device__ __forceinline__ size_t aswz(int m, int k, int K) {
  int mt = m >> 4, l15 = m & 15;
  int kt = k >> 5, kk = k & 31;
  int hi = kk >> 4, e = kk & 15;
  return ((((size_t)mt * (K >> 5) + kt) << 5) + (hi << 4) + l15) * 16 + e;
}
__device__ __forceinline__ size_t bswz(int k, int n, int N) {
  int nt = n >> 4, l15 = n & 15;
  int kt = k >> 5, kk = k & 31;
  int hi = kk >> 4, e = kk & 15;
  return ((((size_t)kt * (N >> 4) + nt) << 5) + (hi << 4) + l15) * 16 + e;
}

// ---------------------------------------------------------------------------
// LDS-free WMMA GEMM on fragment-ordered operands.
// C(MxN, f32 row-major) = A * B.  M%256==0, N%128==0, K%32==0.
// 256 threads = 8 wave32s (4 wave-rows x 2 wave-cols).
// Block tile 256x128; wave tile 64x64 = 4x4 fragments, 16 WMMA per K-step.
// 8 fragment loads feed 16 WMMAs -> 0.5KB operand traffic per WMMA, all
// coalesced global_load_b128 served from the 192MB L2.
// ---------------------------------------------------------------------------
__global__ __launch_bounds__(256) void wmma_gemm_swz(
    const unsigned short* __restrict__ A,
    const unsigned short* __restrict__ B,
    float* __restrict__ C, int M, int N, int K)
{
  const int lane = threadIdx.x & 31;
  const int wid  = threadIdx.x >> 5;
  const int wy   = wid & 3;          // 4 wave-rows (64 rows each)
  const int wx   = wid >> 2;         // 2 wave-cols (64 cols each)
  const int l15  = lane & 15;
  const int hi   = lane >> 4;
  const int KT   = K >> 5;
  const int NT   = N >> 4;

  v8f acc[4][4];
  #pragma unroll
  for (int mt = 0; mt < 4; ++mt)
    #pragma unroll
    for (int nt = 0; nt < 4; ++nt)
      #pragma unroll
      for (int i = 0; i < 8; ++i) acc[mt][nt][i] = 0.f;

  const unsigned short* aP[4];
  #pragma unroll
  for (int mt = 0; mt < 4; ++mt) {
    int mtile = blockIdx.y * 16 + wy * 4 + mt;
    aP[mt] = A + ((size_t)mtile * KT * 32 + lane) * 16;
  }
  const unsigned short* bP[4];
  #pragma unroll
  for (int nt = 0; nt < 4; ++nt) {
    int ntile = blockIdx.x * 8 + wx * 4 + nt;
    bP[nt] = B + ((size_t)ntile * 32 + lane) * 16;
  }
  const size_t aStep = 32 * 16;                 // ushorts per A kt-step
  const size_t bStep = (size_t)NT * 32 * 16;    // ushorts per B kt-step

  for (int kt = 0; kt < KT; ++kt) {
    union { v16u u; uint4 q[2]; } fa[4], fb[4];
    #pragma unroll
    for (int mt = 0; mt < 4; ++mt) {
      const uint4* p = (const uint4*)(aP[mt] + (size_t)kt * aStep);
      fa[mt].q[0] = p[0]; fa[mt].q[1] = p[1];
    }
    #pragma unroll
    for (int nt = 0; nt < 4; ++nt) {
      const uint4* p = (const uint4*)(bP[nt] + (size_t)kt * bStep);
      fb[nt].q[0] = p[0]; fb[nt].q[1] = p[1];
    }
    if (kt + 1 < KT) {   // global_prefetch next K-step chunks toward L0/L2
      #pragma unroll
      for (int mt = 0; mt < 4; ++mt)
        __builtin_prefetch(aP[mt] + (size_t)(kt + 1) * aStep, 0, 1);
      __builtin_prefetch(bP[0] + (size_t)(kt + 1) * bStep, 0, 1);
      __builtin_prefetch(bP[2] + (size_t)(kt + 1) * bStep, 0, 1);
    }
    #pragma unroll
    for (int mt = 0; mt < 4; ++mt)
      #pragma unroll
      for (int nt = 0; nt < 4; ++nt)
        acc[mt][nt] = __builtin_amdgcn_wmma_f32_16x16x32_bf16(
            false, __builtin_bit_cast(v16bf, fa[mt].u),
            false, __builtin_bit_cast(v16bf, fb[nt].u),
            (short)0, acc[mt][nt], false, false);
  }

  #pragma unroll
  for (int mt = 0; mt < 4; ++mt)
    #pragma unroll
    for (int nt = 0; nt < 4; ++nt) {
      const size_t col = (size_t)blockIdx.x * 128 + wx * 64 + nt * 16 + l15;
      #pragma unroll
      for (int v = 0; v < 8; ++v) {
        const size_t row = (size_t)blockIdx.y * 256 + wy * 64 + mt * 16 + v + hi * 8;
        C[row * N + col] = acc[mt][nt][v];
      }
    }
}

// ---------------------------------------------------------------------------
// Producers (write bf16 operands directly in fragment order)
// ---------------------------------------------------------------------------

// weights: src = W (N rows x K cols, f32); B-operand element B[k][n] = W[n][k]
__global__ void pack_w_swz(const float* __restrict__ src,
                           unsigned short* __restrict__ dst, int K, int N) {
  int idx = blockIdx.x * 256 + threadIdx.x;
  if (idx >= K * N) return;
  int n = idx / K, k = idx % K;                 // read coalesced along k
  dst[bswz(k, n, N)] = f2bf(src[idx]);
}

// activations: src (M x K, f32 row-major) -> A fragment order
__global__ void pack_a_swz(const float* __restrict__ src,
                           unsigned short* __restrict__ dst, int M, int K) {
  int idx = blockIdx.x * 256 + threadIdx.x;
  if (idx >= M * K) return;
  int m = idx / K, k = idx % K;
  dst[aswz(m, k, K)] = f2bf(src[idx]);
}

// Forward DFT basis, B-operand (K=SEQL rows t, N=NF2P cols):
// cols [0,513): cos(2pi f t/L); [576,1089): -sin; pads 0. Indexed by output.
__global__ void build_basisF(unsigned short* __restrict__ bF) {
  int idx = blockIdx.x * 256 + threadIdx.x;
  if (idx >= SEQL * NF2P) return;
  int e = idx & 15, lane = (idx >> 4) & 31, tile = idx >> 9;
  int kt = tile / (NF2P >> 4), ntile = tile % (NF2P >> 4);
  int t = kt * 32 + (lane >> 4) * 16 + e;       // K index
  int c = ntile * 16 + (lane & 15);             // N index
  float v = 0.f;
  if (c < NFFT) {
    float ang = 6.28318530717958647692f * (float)((c * t) & (SEQL - 1)) / (float)SEQL;
    v = cosf(ang);
  } else if (c >= FHALF && c < FHALF + NFFT) {
    int f = c - FHALF;
    float ang = 6.28318530717958647692f * (float)((f * t) & (SEQL - 1)) / (float)SEQL;
    v = -sinf(ang);
  }
  bF[idx] = f2bf(v);
}

// Inverse basis, B-operand (K=NF2P rows, N=SEQL cols t):
// rows f<513: wf/L*cos ; rows FHALF+f: -wf/L*sin ; pads 0.
__global__ void build_basisI(unsigned short* __restrict__ bI) {
  int idx = blockIdx.x * 256 + threadIdx.x;
  if (idx >= NF2P * SEQL) return;
  int e = idx & 15, lane = (idx >> 4) & 31, tile = idx >> 9;
  int kt = tile / (SEQL >> 4), ntile = tile % (SEQL >> 4);
  int r = kt * 32 + (lane >> 4) * 16 + e;       // K index
  int t = ntile * 16 + (lane & 15);             // N index
  float v = 0.f;
  if (r < NFFT) {
    float wf = (r == 0 || r == NFFT - 1) ? 1.f : 2.f;
    float ang = 6.28318530717958647692f * (float)((r * t) & (SEQL - 1)) / (float)SEQL;
    v = wf * (1.f / (float)SEQL) * cosf(ang);
  } else if (r >= FHALF && r < FHALF + NFFT) {
    int f = r - FHALF;
    float wf = (f == 0 || f == NFFT - 1) ? 1.f : 2.f;
    float ang = 6.28318530717958647692f * (float)((f * t) & (SEQL - 1)) / (float)SEQL;
    v = -wf * (1.f / (float)SEQL) * sinf(ang);
  }
  bI[idx] = f2bf(v);
}

// split xz -> x_inner (f32), z=silu (f32), transposed A-operand for DFT GEMM
__global__ void split_kernel(const float* __restrict__ xz,
                             float* __restrict__ x_inner, float* __restrict__ zb,
                             unsigned short* __restrict__ xT) {
  int idx = blockIdx.x * 256 + threadIdx.x;
  if (idx >= BL * INNER) return;
  int row = idx >> 11, e = idx & (INNER - 1);
  float xi = xz[(size_t)row * NI2 + e];
  float zi = xz[(size_t)row * NI2 + INNER + e];
  x_inner[idx] = xi;
  zb[idx] = silu(zi);
  int b = row >> 10, t = row & (SEQL - 1);
  xT[aswz(b * INNER + e, t, SEQL)] = f2bf(xi);
}

// depthwise causal conv (K=4) + silu
__global__ void conv_kernel(const float* __restrict__ x_inner,
                            const float* __restrict__ w, const float* __restrict__ bias,
                            float* __restrict__ x_conv) {
  int idx = blockIdx.x * 256 + threadIdx.x;
  if (idx >= BL * INNER) return;
  int row = idx >> 11, e = idx & (INNER - 1);
  int t = row & (SEQL - 1), b = row >> 10;
  const float* xc = x_inner + ((size_t)b * SEQL) * INNER + e;
  float acc = bias[e];
  #pragma unroll
  for (int j = 0; j < CONVK; ++j) {
    int tt = t - (CONVK - 1) + j;
    if (tt >= 0) acc += w[e * CONVK + j] * xc[(size_t)tt * INNER];
  }
  x_conv[idx] = silu(acc);
}

// x_dbc (BL x 40) = x_conv @ x_proj_w^T (tiny: plain VALU dot products)
__global__ void xproj_kernel(const float* __restrict__ xc,
                             const float* __restrict__ w, float* __restrict__ out) {
  int idx = blockIdx.x * 256 + threadIdx.x;
  if (idx >= BL * (DTR + 2 * STATE)) return;
  int row = idx / 40, c = idx % 40;
  const float* a = xc + (size_t)row * INNER;
  const float* ww = w + (size_t)c * INNER;
  float s = 0.f;
  for (int i = 0; i < INNER; ++i) s += a[i] * ww[i];
  out[idx] = s;
}

// selective scan: one thread per (b,e); h[16] in registers; 1024 serial steps
__global__ __launch_bounds__(256) void scan_kernel(
    const float* __restrict__ x_dbc, const float* __restrict__ x_conv,
    const float* __restrict__ A_log, const float* __restrict__ Dv,
    const float* __restrict__ dtw, const float* __restrict__ dtb,
    float* __restrict__ y_ssm, unsigned short* __restrict__ fuA) {
  int idx = blockIdx.x * 256 + threadIdx.x;
  if (idx >= BE) return;
  int b = idx >> 11, e = idx & (INNER - 1);
  float Arow[STATE], h[STATE], w8[DTR];
  #pragma unroll
  for (int s = 0; s < STATE; ++s) { Arow[s] = -__expf(A_log[e * STATE + s]); h[s] = 0.f; }
  #pragma unroll
  for (int r = 0; r < DTR; ++r) w8[r] = dtw[e * DTR + r];
  const float bias = dtb[e], Dd = Dv[e];
  for (int t = 0; t < SEQL; ++t) {
    int row = b * SEQL + t;
    const float* db = x_dbc + (size_t)row * 40;        // wave-uniform -> broadcast
    float acc = bias;
    #pragma unroll
    for (int r = 0; r < DTR; ++r) acc += db[r] * w8[r];
    float dt = (acc > 20.f) ? acc : logf(1.f + __expf(acc));   // softplus
    float xv = x_conv[(size_t)row * INNER + e];
    float dtx = dt * xv;
    float y = 0.f;
    #pragma unroll
    for (int s = 0; s < STATE; ++s) {
      float dA = __expf(dt * Arow[s]);
      h[s] = dA * h[s] + dtx * db[DTR + s];
      y += h[s] * db[DTR + STATE + s];
    }
    y += Dd * xv;
    y_ssm[(size_t)row * INNER + e] = y;
    fuA[aswz(row, e, NI2)] = f2bf(y);                  // left half of fusion A
  }
}

// spectral filter in frequency domain; emits A-operand for inverse GEMM
__global__ void filter_kernel(const float* __restrict__ Xf,
                              const float* __restrict__ fr, const float* __restrict__ fi,
                              const float* __restrict__ decay,
                              unsigned short* __restrict__ Ybf) {
  int idx = blockIdx.x * 256 + threadIdx.x;
  if (idx >= BE * FHALF) return;
  int row = idx / FHALF, c = idx % FHALF;
  if (c >= NFFT) {
    Ybf[aswz(row, c, NF2P)] = 0; Ybf[aswz(row, FHALF + c, NF2P)] = 0; return;
  }
  int e = row & (INNER - 1);
  size_t base = (size_t)row * NF2P;
  float dc = __expf(-decay[e] * ((float)c * (1.f / (float)(NFFT - 1))));
  float Fr = fr[(size_t)e * NFFT + c] * dc;
  float Fi = fi[(size_t)e * NFFT + c] * dc;
  float Xr = Xf[base + c], Xi = Xf[base + FHALF + c];
  Ybf[aswz(row, c, NF2P)]         = f2bf(Xr * Fr - Xi * Fi);
  Ybf[aswz(row, FHALF + c, NF2P)] = f2bf(Xr * Fi + Xi * Fr);
}

// transpose spectral output (b,e,t)->(b,t,e); bf16 right half of fusion A
__global__ void spec_tr_kernel(const float* __restrict__ ysT,
                               float* __restrict__ y_spec, unsigned short* __restrict__ fuA) {
  int idx = blockIdx.x * 256 + threadIdx.x;
  if (idx >= BL * INNER) return;
  int row = idx >> 11, e = idx & (INNER - 1);
  int b = row >> 10, t = row & (SEQL - 1);
  float v = ysT[((size_t)(b * INNER + e)) * SEQL + t];
  y_spec[idx] = v;
  fuA[aswz(row, INNER + e, NI2)] = f2bf(v);
}

// gate + mix + z-gating; produces A-operand for out_proj GEMM
__global__ void combine_kernel(const float* __restrict__ glog, const float* __restrict__ fb,
                               const float* __restrict__ y_ssm, const float* __restrict__ y_spec,
                               const float* __restrict__ zb, unsigned short* __restrict__ ybf) {
  int idx = blockIdx.x * 256 + threadIdx.x;
  if (idx >= BL * INNER) return;
  int row = idx >> 11, e = idx & (INNER - 1);
  float g = 1.f / (1.f + __expf(-(glog[idx] + fb[e])));
  float y = g * y_ssm[idx] + (1.f - g) * y_spec[idx];
  ybf[aswz(row, e, INNER)] = f2bf(y * zb[idx]);
}

// residual add + layernorm: one 256-thread block per row
__global__ __launch_bounds__(256) void ln_kernel(const float* __restrict__ outb,
                                                 const float* __restrict__ resid,
                                                 const float* __restrict__ gamma,
                                                 const float* __restrict__ beta,
                                                 float* __restrict__ out) {
  int row = blockIdx.x;
  __shared__ float ssum[256], ssq[256];
  float v[4]; float s = 0.f, q = 0.f;
  #pragma unroll
  for (int i = 0; i < 4; ++i) {
    int d = threadIdx.x + i * 256;
    float t = outb[(size_t)row * DMODEL + d] + resid[(size_t)row * DMODEL + d];
    v[i] = t; s += t; q += t * t;
  }
  ssum[threadIdx.x] = s; ssq[threadIdx.x] = q; __syncthreads();
  for (int off = 128; off; off >>= 1) {
    if (threadIdx.x < off) { ssum[threadIdx.x] += ssum[threadIdx.x + off];
                             ssq[threadIdx.x]  += ssq[threadIdx.x + off]; }
    __syncthreads();
  }
  float mu = ssum[0] * (1.f / DMODEL);
  float var = ssq[0] * (1.f / DMODEL) - mu * mu;
  float inv = rsqrtf(var + 1e-5f);
  #pragma unroll
  for (int i = 0; i < 4; ++i) {
    int d = threadIdx.x + i * 256;
    out[(size_t)row * DMODEL + d] = (v[i] - mu) * inv * gamma[d] + beta[d];
  }
}

// ---------------------------------------------------------------------------
// Launch
// ---------------------------------------------------------------------------
extern "C" void kernel_launch(void* const* d_in, const int* in_sizes, int n_in,
                              void* d_out, int out_size, void* d_ws, size_t ws_size,
                              hipStream_t stream) {
  (void)in_sizes; (void)n_in; (void)out_size; (void)ws_size;
  const float* x        = (const float*)d_in[0];
  const float* in_w     = (const float*)d_in[1];
  const float* conv_w   = (const float*)d_in[2];
  const float* conv_b   = (const float*)d_in[3];
  const float* A_log    = (const float*)d_in[4];
  const float* Dv       = (const float*)d_in[5];
  const float* xproj_w  = (const float*)d_in[6];
  const float* dtw      = (const float*)d_in[7];
  const float* dtb      = (const float*)d_in[8];
  const float* filt_r   = (const float*)d_in[9];
  const float* filt_i   = (const float*)d_in[10];
  const float* sdecay   = (const float*)d_in[11];
  const float* fus_w    = (const float*)d_in[12];
  const float* fus_b    = (const float*)d_in[13];
  const float* out_w    = (const float*)d_in[14];
  const float* ln_g     = (const float*)d_in[15];
  const float* ln_b     = (const float*)d_in[16];
  float* out = (float*)d_out;

  // ---- workspace bump allocator (~215 MB) ----
  char* w = (char*)d_ws; size_t off = 0;
  auto alloc = [&](size_t bytes) -> void* {
    void* p = w + off; off = (off + bytes + 255) & ~(size_t)255; return p;
  };
  unsigned short* wInB   = (unsigned short*)alloc((size_t)DMODEL * NI2 * 2);
  unsigned short* wFuB   = (unsigned short*)alloc((size_t)NI2 * INNER * 2);
  unsigned short* wOutB  = (unsigned short*)alloc((size_t)INNER * DMODEL * 2);
  unsigned short* basisF = (unsigned short*)alloc((size_t)SEQL * NF2P * 2);
  unsigned short* basisI = (unsigned short*)alloc((size_t)NF2P * SEQL * 2);
  unsigned short* xbf    = (unsigned short*)alloc((size_t)BL * DMODEL * 2);
  unsigned short* xTbf   = (unsigned short*)alloc((size_t)BE * SEQL * 2);
  float* x_inner = (float*)alloc((size_t)BL * INNER * 4);
  float* zb      = (float*)alloc((size_t)BL * INNER * 4);
  float* x_conv  = (float*)alloc((size_t)BL * INNER * 4);
  float* x_dbc   = (float*)alloc((size_t)BL * 40 * 4);
  float* y_ssm   = (float*)alloc((size_t)BL * INNER * 4);
  float* y_spec  = (float*)alloc((size_t)BL * INNER * 4);
  unsigned short* fuA = (unsigned short*)alloc((size_t)BL * NI2 * 2);
  float* glog    = (float*)alloc((size_t)BL * INNER * 4);
  unsigned short* ybf = (unsigned short*)alloc((size_t)BL * INNER * 2);
  float* outb    = (float*)alloc((size_t)BL * DMODEL * 4);
  float* xz      = (float*)alloc((size_t)BL * NI2 * 4);          // 33.5 MB
  // aliases inside the xz region (dead after split_kernel):
  float*          Xf  = xz;                                                    // 18.9 MB
  unsigned short* Ybf = (unsigned short*)((char*)xz + (size_t)BE * NF2P * 4);  // +9.4 MB
  // x_inner region is dead after conv_kernel:
  float* ySpecT = x_inner;                                       // BE x SEQL f32

  dim3 blk(256);
  auto g1 = [](size_t n) { return dim3((unsigned)((n + 255) / 256)); };

  // ---- operand packing + basis generation (fragment order) ----
  pack_w_swz<<<g1((size_t)NI2 * DMODEL), blk, 0, stream>>>(in_w,  wInB,  DMODEL, NI2);
  pack_w_swz<<<g1((size_t)INNER * NI2),  blk, 0, stream>>>(fus_w, wFuB,  NI2,    INNER);
  pack_w_swz<<<g1((size_t)DMODEL*INNER), blk, 0, stream>>>(out_w, wOutB, INNER,  DMODEL);
  build_basisF<<<g1((size_t)SEQL * NF2P), blk, 0, stream>>>(basisF);
  build_basisI<<<g1((size_t)NF2P * SEQL), blk, 0, stream>>>(basisI);
  pack_a_swz<<<g1((size_t)BL * DMODEL), blk, 0, stream>>>(x, xbf, BL, DMODEL);

  // ---- in_proj GEMM: (2048x1024)@(1024x4096) -> xz ----
  wmma_gemm_swz<<<dim3(NI2 / 128, BL / 256), blk, 0, stream>>>(xbf, wInB, xz, BL, NI2, DMODEL);
  split_kernel<<<g1((size_t)BL * INNER), blk, 0, stream>>>(xz, x_inner, zb, xTbf);

  // ---- depthwise conv + silu; x_proj ----
  conv_kernel<<<g1((size_t)BL * INNER), blk, 0, stream>>>(x_inner, conv_w, conv_b, x_conv);
  xproj_kernel<<<g1((size_t)BL * 40), blk, 0, stream>>>(x_conv, xproj_w, x_dbc);

  // ---- selective scan (register-resident state, 4096 parallel chains) ----
  scan_kernel<<<g1(BE), blk, 0, stream>>>(x_dbc, x_conv, A_log, Dv, dtw, dtb, y_ssm, fuA);

  // ---- spectral branch as two WMMA GEMMs (DFT bases) ----
  wmma_gemm_swz<<<dim3(NF2P / 128, BE / 256), blk, 0, stream>>>(xTbf, basisF, Xf, BE, NF2P, SEQL);
  filter_kernel<<<g1((size_t)BE * FHALF), blk, 0, stream>>>(Xf, filt_r, filt_i, sdecay, Ybf);
  wmma_gemm_swz<<<dim3(SEQL / 128, BE / 256), blk, 0, stream>>>(Ybf, basisI, ySpecT, BE, SEQL, NF2P);
  spec_tr_kernel<<<g1((size_t)BL * INNER), blk, 0, stream>>>(ySpecT, y_spec, fuA);

  // ---- fusion gate GEMM + mix + z-gate ----
  wmma_gemm_swz<<<dim3(INNER / 128, BL / 256), blk, 0, stream>>>(fuA, wFuB, glog, BL, INNER, NI2);
  combine_kernel<<<g1((size_t)BL * INNER), blk, 0, stream>>>(glog, fus_b, y_ssm, y_spec, zb, ybf);

  // ---- out_proj GEMM + residual + layernorm ----
  wmma_gemm_swz<<<dim3(DMODEL / 128, BL / 256), blk, 0, stream>>>(ybf, wOutB, outb, BL, DMODEL, INNER);
  ln_kernel<<<dim3(BL), blk, 0, stream>>>(outb, x, ln_g, ln_b, out);
}
